// ECCPagedAttentionShim_34780645163420
// MI455X (gfx1250) — matrix operand order repaired and checked by reading, not compiled, not run on hardware
//
#include <hip/hip_runtime.h>

// ---------------- problem constants ----------------
#define B_  2
#define S_  2048
#define DM_ 4096
#define H_  32
#define HKV_ 8
#define D_  128
#define MROWS (B_*S_)          // 4096
#define QN (H_*D_)             // 4096
#define KN (HKV_*D_)           // 1024

typedef __attribute__((ext_vector_type(16))) __bf16 v16bf;
typedef __attribute__((ext_vector_type(8)))  float  v8f;
typedef __attribute__((ext_vector_type(4)))  int    v4i;

typedef __attribute__((address_space(1))) v4i as1_v4i;   // global int4
typedef __attribute__((address_space(3))) v4i as3_v4i;   // LDS int4

#if defined(__has_builtin)
#  if __has_builtin(__builtin_amdgcn_global_load_async_to_lds_b128)
#    define HAVE_ASYNC_LDS 1
#  else
#    define HAVE_ASYNC_LDS 0
#  endif
#  if __has_builtin(__builtin_amdgcn_s_wait_asynccnt)
#    define HAVE_WAIT_ASYNC 1
#  else
#    define HAVE_WAIT_ASYNC 0
#  endif
#else
#  define HAVE_ASYNC_LDS 0
#  define HAVE_WAIT_ASYNC 0
#endif

union BFU { unsigned short u; __bf16 b; };

__device__ __forceinline__ unsigned short f2bf_bits(float f) {
    union { float f; unsigned int u; } c; c.f = f;
    unsigned int u = c.u;
    u += 0x7FFFu + ((u >> 16) & 1u);   // round-to-nearest-even
    return (unsigned short)(u >> 16);
}
__device__ __forceinline__ __bf16 us2bf(unsigned short u) { BFU x; x.u = u; return x.b; }

__device__ __forceinline__ void async_wait0() {
#if HAVE_ASYNC_LDS
#  if HAVE_WAIT_ASYNC
    __builtin_amdgcn_s_wait_asynccnt(0);
#  else
    asm volatile("s_wait_asynccnt 0x0" ::: "memory");
#  endif
#endif
}

// =====================================================================
// GEMM: C[m,n] = sum_k A[m,k] * W[n,k]   (A: f32 or bf16-bits, W: f32)
// Macro tile 128(M) x 128(N), BK=32. 256 threads = 8 waves in 4x2 grid,
// each wave owns a 32x64 block = 2x4 quad of 16x16 WMMA tiles
// (8 WMMAs per wave per K-chunk). global_prefetch for the next chunk.
// =====================================================================
template <bool AF32>
__global__ void __launch_bounds__(256)
wmma_gemm_xwt(const void* __restrict__ Ap, const float* __restrict__ W,
              float* __restrict__ C, int M, int N, int K)
{
    __shared__ unsigned short As[128][34];
    __shared__ unsigned short Bs[128][34];

    const int m0 = blockIdx.y * 128;
    const int n0 = blockIdx.x * 128;
    const int wid  = threadIdx.x >> 5;
    const int lane = threadIdx.x & 31;
    const int wm = wid & 3, wn = wid >> 2;
    const int half = lane >> 4, ln = lane & 15;

    v8f acc[2][4] = {};

    for (int k0 = 0; k0 < K; k0 += 32) {
        // prefetch next K-chunk rows (one row per thread) -> global_prefetch_b8
        if (k0 + 32 < K) {
            if (threadIdx.x < 128) {
                if (AF32)
                    __builtin_prefetch(&((const float*)Ap)[(size_t)(m0 + threadIdx.x) * K + k0 + 32], 0, 3);
                else
                    __builtin_prefetch(&((const unsigned short*)Ap)[(size_t)(m0 + threadIdx.x) * K + k0 + 32], 0, 3);
            } else {
                __builtin_prefetch(&W[(size_t)(n0 + threadIdx.x - 128) * K + k0 + 32], 0, 3);
            }
        }
        // stage A tile (convert f32->bf16 if needed)
#pragma unroll
        for (int it = 0; it < 16; ++it) {
            int i = it * 256 + threadIdx.x;
            int r = i >> 5, c = i & 31;
            unsigned short v;
            if (AF32) v = f2bf_bits(((const float*)Ap)[(size_t)(m0 + r) * K + k0 + c]);
            else      v = ((const unsigned short*)Ap)[(size_t)(m0 + r) * K + k0 + c];
            As[r][c] = v;
        }
        // stage B tile from W[n][k]
#pragma unroll
        for (int it = 0; it < 16; ++it) {
            int i = it * 256 + threadIdx.x;
            int r = i >> 5, c = i & 31;
            Bs[r][c] = f2bf_bits(W[(size_t)(n0 + r) * K + k0 + c]);
        }
        __syncthreads();

        v16bf af[2], bfr[4];
#pragma unroll
        for (int i = 0; i < 2; ++i) {
            int row = wm * 32 + i * 16 + ln;
#pragma unroll
            for (int e = 0; e < 16; ++e) {
                int kk = e + ((e >= 8) ? 8 : 0) + (half ? 8 : 0);   // A-frag K map
                af[i][e] = us2bf(As[row][kk]);
            }
        }
#pragma unroll
        for (int j = 0; j < 4; ++j) {
            int col = wn * 64 + j * 16 + ln;
#pragma unroll
            for (int e = 0; e < 16; ++e) {
                int kk = e + ((e >= 8) ? 8 : 0) + (half ? 8 : 0);   // B-frag K map
                bfr[j][e] = us2bf(Bs[col][kk]);
            }
        }
#pragma unroll
        for (int i = 0; i < 2; ++i)
#pragma unroll
            for (int j = 0; j < 4; ++j)
                acc[i][j] = __builtin_amdgcn_wmma_f32_16x16x32_bf16(
                    false, af[i], false, bfr[j], (short)0, acc[i][j], false, false);
        __syncthreads();
    }

#pragma unroll
    for (int i = 0; i < 2; ++i)
#pragma unroll
        for (int j = 0; j < 4; ++j)
#pragma unroll
            for (int r = 0; r < 8; ++r) {
                int row = m0 + wm * 32 + i * 16 + r + half * 8;
                int col = n0 + wn * 64 + j * 16 + ln;
                C[(size_t)row * N + col] = acc[i][j][r];
            }
}

// =====================================================================
// RoPE + pack to [b, h, s, d] bf16
// =====================================================================
__global__ void __launch_bounds__(256)
rope_pack(const float* __restrict__ src, const int* __restrict__ pos,
          unsigned short* __restrict__ dst, int nheads)
{
    int i = blockIdx.x * blockDim.x + threadIdx.x;          // total B*S*nheads*64
    int d = i & 63;
    int h = (i >> 6) % nheads;
    int rest = i / (64 * nheads);
    int s = rest & (S_ - 1);
    int b = rest >> 11;

    float p = (float)pos[b * S_ + s];
    float inv = __powf(10000.0f, -(float)d * (1.0f / 64.0f));
    float ang = p * inv;
    float sn, cs; __sincosf(ang, &sn, &cs);

    const float* row = src + (size_t)(b * S_ + s) * (nheads * D_) + h * D_;
    float x1 = row[d], x2 = row[d + 64];
    size_t o = (((size_t)(b * nheads + h)) * S_ + s) * D_ + d;
    dst[o]      = f2bf_bits(x1 * cs - x2 * sn);
    dst[o + 64] = f2bf_bits(x2 * cs + x1 * sn);
}

__global__ void __launch_bounds__(256)
pack_v(const float* __restrict__ src, unsigned short* __restrict__ dst)
{
    int i = blockIdx.x * blockDim.x + threadIdx.x;          // total B*S*HKV*128
    int d  = i & 127;
    int hk = (i >> 7) & 7;
    int s  = (i >> 10) & (S_ - 1);
    int b  = i >> 21;
    dst[(((size_t)(b * HKV_ + hk)) * S_ + s) * D_ + d] =
        f2bf_bits(src[(size_t)(b * S_ + s) * KN + hk * D_ + d]);
}

// =====================================================================
// Flash attention: block = 128 threads (4 waves), each wave owns a 16-row
// Q tile; block covers 64 query rows. K/V staged in LDS as 32x128 bf16
// via GLOBAL_LOAD_ASYNC_TO_LDS_B128 (ASYNCcnt) when available.
// =====================================================================
__global__ void __launch_bounds__(128)
flash_attn(const unsigned short* __restrict__ qb,
           const unsigned short* __restrict__ kb,
           const unsigned short* __restrict__ vb,
           unsigned short* __restrict__ ab)
{
    const int qblk = blockIdx.x, h = blockIdx.y, b = blockIdx.z;
    const int hkv = h >> 2;                                  // H/HKV = 4
    const int wid  = threadIdx.x >> 5;
    const int lane = threadIdx.x & 31;
    const int half = lane >> 4, ln = lane & 15;
    const int q0 = qblk * 64 + wid * 16;

    __shared__ unsigned short Ks[32][136];
    __shared__ unsigned short Vs[32][136];
    __shared__ unsigned short Ps[4][16][34];

    // ---- load Q fragments (A layout), D=128 -> 4 chunks of K=32 ----
    const unsigned short* Q = qb + ((size_t)(b * H_ + h)) * S_ * D_;
    v16bf qfrag[4];
    {
        int m = q0 + ln;
#pragma unroll
        for (int c = 0; c < 4; ++c)
#pragma unroll
            for (int e = 0; e < 16; ++e) {
                int kk = c * 32 + e + ((e >= 8) ? 8 : 0) + (half ? 8 : 0);
                qfrag[c][e] = us2bf(Q[(size_t)m * D_ + kk]);
            }
    }

    v8f acc[8] = {};
    float m_i[8], l_i[8];
#pragma unroll
    for (int r = 0; r < 8; ++r) { m_i[r] = -1e30f; l_i[r] = 0.0f; }

    const float scale = 0.08838834764831845f;                // 1/sqrt(128)
    const unsigned short* Kb = kb + ((size_t)(b * HKV_ + hkv)) * S_ * D_;
    const unsigned short* Vb = vb + ((size_t)(b * HKV_ + hkv)) * S_ * D_;

    const int kend = qblk * 64 + 64;                         // causal bound
    for (int k0 = 0; k0 < kend; k0 += 32) {
#if HAVE_ASYNC_LDS
        // async global->LDS copies: 16B per lane, 4 issues/thread per array
        for (int i = threadIdx.x; i < 512; i += 128) {
            int r = i >> 4, c8 = (i & 15) * 8;               // 8 ushorts = 16B
            __builtin_amdgcn_global_load_async_to_lds_b128(
                (as1_v4i*)(uintptr_t)&Kb[(size_t)(k0 + r) * D_ + c8],
                (as3_v4i*)(unsigned int)(uintptr_t)&Ks[r][c8],
                0, 0);
            __builtin_amdgcn_global_load_async_to_lds_b128(
                (as1_v4i*)(uintptr_t)&Vb[(size_t)(k0 + r) * D_ + c8],
                (as3_v4i*)(unsigned int)(uintptr_t)&Vs[r][c8],
                0, 0);
        }
        async_wait0();
#else
        for (int i = threadIdx.x; i < 32 * 32; i += 128) {
            int r = i >> 5, d4 = (i & 31) * 4;
            *(uint2*)&Ks[r][d4] = *(const uint2*)&Kb[(size_t)(k0 + r) * D_ + d4];
            *(uint2*)&Vs[r][d4] = *(const uint2*)&Vb[(size_t)(k0 + r) * D_ + d4];
        }
#endif
        __syncthreads();

        // ---- scores: two 16-key tiles, 4 WMMAs each over D ----
        v8f sc[2] = {};
#pragma unroll
        for (int j = 0; j < 2; ++j)
#pragma unroll
            for (int c = 0; c < 4; ++c) {
                v16bf kf;
#pragma unroll
                for (int e = 0; e < 16; ++e) {
                    int dd = c * 32 + e + ((e >= 8) ? 8 : 0) + (half ? 8 : 0);   // B-frag K map
                    kf[e] = us2bf(Ks[j * 16 + ln][dd]);
                }
                sc[j] = __builtin_amdgcn_wmma_f32_16x16x32_bf16(
                    false, qfrag[c], false, kf, (short)0, sc[j], false, false);
            }

        // ---- causal mask + online softmax (rows across 16-lane halves) ----
        float sv[2][8];
#pragma unroll
        for (int j = 0; j < 2; ++j) {
            int kcol = k0 + j * 16 + ln;
#pragma unroll
            for (int r = 0; r < 8; ++r) {
                int qrow = q0 + r + half * 8;
                float v = sc[j][r] * scale;
                sv[j][r] = (kcol <= qrow) ? v : -1e30f;
            }
        }
        float tmax[8], rsum[8];
#pragma unroll
        for (int r = 0; r < 8; ++r) {
            tmax[r] = fmaxf(sv[0][r], sv[1][r]);
#pragma unroll
            for (int mk = 1; mk < 16; mk <<= 1)
                tmax[r] = fmaxf(tmax[r], __shfl_xor(tmax[r], mk, 16));
        }
        float mnew[8], alpha[8];
#pragma unroll
        for (int r = 0; r < 8; ++r) {
            mnew[r]  = fmaxf(m_i[r], tmax[r]);
            alpha[r] = __expf(m_i[r] - mnew[r]);
            sv[0][r] = __expf(sv[0][r] - mnew[r]);
            sv[1][r] = __expf(sv[1][r] - mnew[r]);
            rsum[r]  = sv[0][r] + sv[1][r];
        }
#pragma unroll
        for (int r = 0; r < 8; ++r) {
#pragma unroll
            for (int mk = 1; mk < 16; mk <<= 1)
                rsum[r] += __shfl_xor(rsum[r], mk, 16);
            l_i[r] = l_i[r] * alpha[r] + rsum[r];
            m_i[r] = mnew[r];
        }
#pragma unroll
        for (int t = 0; t < 8; ++t)
#pragma unroll
            for (int r = 0; r < 8; ++r)
                acc[t][r] *= alpha[r];

        // ---- C-layout P -> LDS -> A-fragment layout (per-wave patch) ----
#pragma unroll
        for (int r = 0; r < 8; ++r) {
            Ps[wid][r + half * 8][ln]      = f2bf_bits(sv[0][r]);
            Ps[wid][r + half * 8][16 + ln] = f2bf_bits(sv[1][r]);
        }
        v16bf pfrag;
#pragma unroll
        for (int e = 0; e < 16; ++e) {
            int kk = e + ((e >= 8) ? 8 : 0) + (half ? 8 : 0);
            pfrag[e] = us2bf(Ps[wid][ln][kk]);
        }

        // ---- PV: 8 N-tiles of 16 over D=128 ----
#pragma unroll
        for (int t = 0; t < 8; ++t) {
            v16bf vf;
#pragma unroll
            for (int e = 0; e < 16; ++e) {
                int key = e + ((e >= 8) ? 8 : 0) + (half ? 8 : 0);   // B-frag K map
                vf[e] = us2bf(Vs[key][t * 16 + ln]);
            }
            acc[t] = __builtin_amdgcn_wmma_f32_16x16x32_bf16(
                false, pfrag, false, vf, (short)0, acc[t], false, false);
        }
        __syncthreads();
    }

    // ---- normalize + store attn out as bf16 in (B*S, H*D) layout ----
    float inv_l[8];
#pragma unroll
    for (int r = 0; r < 8; ++r) inv_l[r] = 1.0f / l_i[r];
#pragma unroll
    for (int t = 0; t < 8; ++t)
#pragma unroll
        for (int r = 0; r < 8; ++r) {
            int row = q0 + r + half * 8;
            size_t o = ((size_t)(b * S_ + row)) * QN + h * D_ + t * 16 + ln;
            ab[o] = f2bf_bits(acc[t][r] * inv_l[r]);
        }
}

// =====================================================================
extern "C" void kernel_launch(void* const* d_in, const int* in_sizes, int n_in,
                              void* d_out, int out_size, void* d_ws, size_t ws_size,
                              hipStream_t stream) {
    const float* X   = (const float*)d_in[0];   // (B,S,DM) f32
    const int*   pos = (const int*)  d_in[1];   // (B,S) i32
    const float* Wq  = (const float*)d_in[2];   // (4096,4096)
    const float* Wk  = (const float*)d_in[3];   // (1024,4096)
    const float* Wv  = (const float*)d_in[4];   // (1024,4096)
    const float* Wo  = (const float*)d_in[5];   // (4096,4096)
    float* out = (float*)d_out;                 // (B,S,4096) f32

    // workspace layout (184.6 MB total)
    char* w = (char*)d_ws;
    float* qf = (float*)w;                    w += (size_t)MROWS * QN * 4;  // 64 MB
    float* kf = (float*)w;                    w += (size_t)MROWS * KN * 4;  // 16 MB
    float* vf = (float*)w;                    w += (size_t)MROWS * KN * 4;  // 16 MB
    unsigned short* qb = (unsigned short*)w;  w += (size_t)MROWS * QN * 2;  // 32 MB
    unsigned short* kb = (unsigned short*)w;  w += (size_t)MROWS * KN * 2;  // 8 MB
    unsigned short* vb = (unsigned short*)w;  w += (size_t)MROWS * KN * 2;  // 8 MB
    unsigned short* ab = (unsigned short*)w;  w += (size_t)MROWS * QN * 2;  // 32 MB

    // 1) QKV projections (f32 A, bf16 WMMA)
    wmma_gemm_xwt<true><<<dim3(QN / 128, MROWS / 128), 256, 0, stream>>>(X, Wq, qf, MROWS, QN, DM_);
    wmma_gemm_xwt<true><<<dim3(KN / 128, MROWS / 128), 256, 0, stream>>>(X, Wk, kf, MROWS, KN, DM_);
    wmma_gemm_xwt<true><<<dim3(KN / 128, MROWS / 128), 256, 0, stream>>>(X, Wv, vf, MROWS, KN, DM_);

    // 2) RoPE + pack to per-head bf16
    rope_pack<<<(B_ * S_ * H_   * 64) / 256, 256, 0, stream>>>(qf, pos, qb, H_);
    rope_pack<<<(B_ * S_ * HKV_ * 64) / 256, 256, 0, stream>>>(kf, pos, kb, HKV_);
    pack_v  <<<(B_ * S_ * HKV_ * 128) / 256, 256, 0, stream>>>(vf, vb);

    // 3) causal flash attention
    flash_attn<<<dim3(S_ / 64, H_, B_), 128, 0, stream>>>(qb, kb, vb, ab);

    // 4) output projection (bf16 A, f32 W) -> f32 out
    wmma_gemm_xwt<false><<<dim3(QN / 128, MROWS / 128), 256, 0, stream>>>(ab, Wo, out, MROWS, QN, QN);
}
